// ResGraphMAE_46978352284510
// MI455X (gfx1250) — matrix operand order repaired and checked by reading, not compiled, not run on hardware
//
#include <hip/hip_runtime.h>
#include <math.h>

#define D 128
#define BN_EPS 1e-5f

typedef __attribute__((ext_vector_type(16))) __bf16 v16bf_t;
typedef __attribute__((ext_vector_type(8)))  float  v8f_t;

__device__ __forceinline__ unsigned short f2bf(float f) {
  union { float f; unsigned int u; } cv; cv.f = f;
  unsigned int u = cv.u;
  unsigned int r = u + 0x7fffu + ((u >> 16) & 1u);   // round-to-nearest-even
  return (unsigned short)(r >> 16);
}

// ---------------- degree / normalization ----------------
__global__ void deg_init(float* __restrict__ deg, int n) {
  int i = blockIdx.x * blockDim.x + threadIdx.x;
  if (i < n) deg[i] = 1.0f;                          // self-loop
}
__global__ void deg_edges(const int* __restrict__ dst, float* __restrict__ deg, int E_) {
  int e = blockIdx.x * blockDim.x + threadIdx.x;
  if (e < E_) atomicAdd(&deg[dst[e]], 1.0f);
}
__global__ void deg_inv(float* __restrict__ deg, int n) {
  int i = blockIdx.x * blockDim.x + threadIdx.x;
  if (i < n) deg[i] = rsqrtf(deg[i]);                // deg >= 1 always
}

// ---------------- mask tokens ----------------
__global__ void mask_enc(const float* __restrict__ x, const int* __restrict__ mv,
                         const float* __restrict__ tok, float* __restrict__ h, int n) {
  int i = blockIdx.x * blockDim.x + threadIdx.x;
  if (i >= n * D) return;
  int row = i >> 7, c = i & (D - 1);
  h[i] = (mv[row] == 0) ? tok[c] : x[i];
}
__global__ void mask_dec(const int* __restrict__ mv, const float* __restrict__ tok,
                         float* __restrict__ h, int n) {
  int i = blockIdx.x * blockDim.x + threadIdx.x;
  if (i >= n * D) return;
  int row = i >> 7, c = i & (D - 1);
  if (mv[row] == 0) h[i] = tok[c];
}

// ---------------- WMMA GEMM: C = A[n,128] @ Bw[128,128] (+bias,+sigmoid) ----------------
// Block: 256 thr (8 waves), 64 rows per block, full 128 cols.
// Wave (w&3) -> 16-row tile, (w>>2) -> 64-col half; 4 x v_wmma_f32_16x16x32_bf16 per K-step.
__global__ __launch_bounds__(256) void gemm_bf16_wmma(
    const float* __restrict__ A, const float* __restrict__ Bw,
    float* __restrict__ Cout, const float* __restrict__ bias, int n, int mode)
{
  __shared__ unsigned short Wl[D * D];          // 32 KB  bf16 W, [k][j] row-major
  __shared__ unsigned short Al[64 * (D + 8)];   // ~17 KB bf16 A tile (padded)
  const int tid = threadIdx.x;
  const int rowbase = blockIdx.x * 64;

  // speculative prefetch of the next block's A tile (gfx1250 global_prefetch_b8)
  __builtin_prefetch(A + (size_t)(rowbase + 64) * D, 0, 1);

  for (int i = tid; i < D * D; i += 256) Wl[i] = f2bf(Bw[i]);
  for (int i = tid; i < 64 * D; i += 256) {
    int r = i >> 7, c = i & (D - 1);
    int row = rowbase + r;
    float v = (row < n) ? A[(size_t)row * D + c] : 0.0f;
    Al[r * (D + 8) + c] = f2bf(v);
  }
  __syncthreads();

  const int wave = tid >> 5;
  const int lane = tid & 31;
  const int mtile = wave & 3;      // which 16-row tile
  const int nhalf = wave >> 2;     // which 64-col half
  const int hi   = lane >> 4;
  const int l15  = lane & 15;

  v8f_t acc[4] = {};

  const unsigned short* Ar = &Al[(mtile * 16 + l15) * (D + 8)];
  for (int k0 = 0; k0 < D; k0 += 32) {
    union { unsigned int u[8]; v16bf_t v; } af;
#pragma unroll
    for (int t = 0; t < 8; ++t) {
      int g = t >> 2, vv = t & 3;
      int k = k0 + g * 16 + hi * 8 + vv * 2;     // A 16-bit 16x32 layout
      af.u[t] = *(const unsigned int*)&Ar[k];
    }
    const unsigned short* Br = &Wl[k0 * D + l15]; // B: lane -> N column
#pragma unroll
    for (int t4 = 0; t4 < 4; ++t4) {
      union { unsigned int u[8]; v16bf_t v; } bf;
#pragma unroll
      for (int t = 0; t < 8; ++t) {
        int nn = nhalf * 64 + t4 * 16;
        int kk = (t >> 2) * 16 + hi * 8 + (t & 3) * 2;   // B: VGPR element -> K pair (same as A)
        bf.u[t] = (unsigned int)Br[kk * D + nn] | ((unsigned int)Br[(kk + 1) * D + nn] << 16);
      }
      acc[t4] = __builtin_amdgcn_wmma_f32_16x16x32_bf16(
          false, af.v, false, bf.v, (short)0, acc[t4], false, false);
    }
  }

#pragma unroll
  for (int t4 = 0; t4 < 4; ++t4) {
    int col = nhalf * 64 + t4 * 16 + l15;        // C/D: N = lane&15
#pragma unroll
    for (int r = 0; r < 8; ++r) {
      int row = rowbase + mtile * 16 + r + hi * 8;  // C/D: M = r + 8*(lane>=16)
      if (row < n) {
        float v = acc[t4][r];
        if (mode) { v += bias[col]; v = 1.0f / (1.0f + expf(-v)); }
        Cout[(size_t)row * D + col] = v;
      }
    }
  }
}

// ---------------- aggregation ----------------
__global__ void agg_init(const float* __restrict__ m, const float* __restrict__ dinv,
                         const float* __restrict__ bvec, float* __restrict__ agg, int n) {
  int i = blockIdx.x * blockDim.x + threadIdx.x;   // float4 index
  if (i >= n * 32) return;
  int row = i >> 5, cb = i & 31;
  float di = dinv[row]; float w = di * di;         // self-loop norm
  float4 v = ((const float4*)m)[i];
  float4 bb = ((const float4*)bvec)[cb];
  float4 o;
  o.x = bb.x + v.x * w; o.y = bb.y + v.y * w;
  o.z = bb.z + v.z * w; o.w = bb.w + v.w * w;
  ((float4*)agg)[i] = o;
}

__global__ void agg_edges(const float* __restrict__ m, const float* __restrict__ dinv,
                          const int* __restrict__ src, const int* __restrict__ dst,
                          float* __restrict__ agg, int E_) {
  int g = blockIdx.x * blockDim.x + threadIdx.x;
  int e = g >> 5;
  if (e >= E_) return;
  int lane = g & 31;
  int s = src[e], d = dst[e];
  float w = dinv[s] * dinv[d];
  float4 v = ((const float4*)(m + (size_t)s * D))[lane];
  float* out = agg + (size_t)d * D + lane * 4;
  atomicAdd(out + 0, v.x * w);
  atomicAdd(out + 1, v.y * w);
  atomicAdd(out + 2, v.z * w);
  atomicAdd(out + 3, v.w * w);
}

// ---------------- batchnorm ----------------
__global__ __launch_bounds__(256) void bn_partial(const float* __restrict__ agg,
                                                  float* __restrict__ stats, int n) {
  __shared__ float ls[256], lq[256];
  int c = threadIdx.x & (D - 1), part = threadIdx.x >> 7;
  float s = 0.f, q = 0.f;
  for (int r = blockIdx.x * 2 + part; r < n; r += gridDim.x * 2) {
    float v = agg[(size_t)r * D + c];
    s += v; q += v * v;
  }
  ls[threadIdx.x] = s; lq[threadIdx.x] = q;
  __syncthreads();
  if (threadIdx.x < D) {
    atomicAdd(&stats[c],     ls[threadIdx.x] + ls[threadIdx.x + D]);
    atomicAdd(&stats[D + c], lq[threadIdx.x] + lq[threadIdx.x + D]);
  }
}

__global__ void bn_finalize(float* __restrict__ stats, const float* __restrict__ gamma,
                            const float* __restrict__ beta, int n) {
  int c = threadIdx.x;
  if (c >= D) return;
  float inv_n = 1.0f / (float)n;
  float mu = stats[c] * inv_n;
  float var = stats[D + c] * inv_n - mu * mu;
  var = var < 0.f ? 0.f : var;
  float sc = gamma[c] * rsqrtf(var + BN_EPS);
  float sh = beta[c] - mu * sc;
  stats[c] = sc; stats[D + c] = sh;
}

__global__ void bn_apply(const float* __restrict__ agg, const float* __restrict__ stats,
                         const float* __restrict__ aP, int layer,
                         float* __restrict__ h, int n) {
  int i = blockIdx.x * blockDim.x + threadIdx.x;   // float4 index
  if (i >= n * 32) return;
  int cb = (i & 31) * 4;
  float slope = aP[layer];
  float4 g = ((const float4*)agg)[i];
  float4 hh = ((float4*)h)[i];
  float4 sc = *(const float4*)&stats[cb];
  float4 sh = *(const float4*)&stats[D + cb];
  float v;
  v = g.x * sc.x + sh.x; hh.x += (v >= 0.f) ? v : slope * v;
  v = g.y * sc.y + sh.y; hh.y += (v >= 0.f) ? v : slope * v;
  v = g.z * sc.z + sh.z; hh.z += (v >= 0.f) ? v : slope * v;
  v = g.w * sc.w + sh.w; hh.w += (v >= 0.f) ? v : slope * v;
  ((float4*)h)[i] = hh;
}

// ---------------- orchestration ----------------
extern "C" void kernel_launch(void* const* d_in, const int* in_sizes, int n_in,
                              void* d_out, int out_size, void* d_ws, size_t ws_size,
                              hipStream_t stream) {
  (void)n_in; (void)out_size; (void)ws_size;
  const float* x   = (const float*)d_in[0];
  const int*   ei  = (const int*)d_in[1];
  const int*   mv  = (const int*)d_in[2];
  const float* enc = (const float*)d_in[3];
  const float* dec = (const float*)d_in[4];
  const float* W   = (const float*)d_in[5];
  const float* b   = (const float*)d_in[6];
  const float* gm  = (const float*)d_in[7];
  const float* bt  = (const float*)d_in[8];
  const float* aP  = (const float*)d_in[9];
  const float* Wp  = (const float*)d_in[10];
  const float* bp  = (const float*)d_in[11];

  const int n  = in_sizes[0] / D;
  const int E_ = in_sizes[1] / 2;
  const int* srcI = ei;
  const int* dstI = ei + E_;

  char* ws = (char*)d_ws;
  const size_t szND = (size_t)n * D * sizeof(float);
  float* h    = (float*)ws;
  float* m    = (float*)(ws + szND);
  float* agg  = (float*)(ws + 2 * szND);
  float* dinv = (float*)(ws + 3 * szND);
  const size_t szN = (((size_t)n * sizeof(float)) + 255) & ~(size_t)255;
  float* stats = (float*)(ws + 3 * szND + szN);

  const int TB = 256;
  dim3 blk(TB);
  const int gN   = (n + TB - 1) / TB;
  const int gE   = (E_ + TB - 1) / TB;
  const int gND  = (n * D + TB - 1) / TB;
  const int gV4  = (n * 32 + TB - 1) / TB;
  const int gEd  = (int)(((size_t)E_ * 32 + TB - 1) / TB);
  const int gGem = (n + 63) / 64;

  deg_init <<<gN, blk, 0, stream>>>(dinv, n);
  deg_edges<<<gE, blk, 0, stream>>>(dstI, dinv, E_);
  deg_inv  <<<gN, blk, 0, stream>>>(dinv, n);

  mask_enc<<<gND, blk, 0, stream>>>(x, mv, enc, h, n);

  for (int layer = 0; layer < 4; ++layer) {
    gemm_bf16_wmma<<<gGem, blk, 0, stream>>>(h, W + (size_t)layer * D * D, m, nullptr, n, 0);
    agg_init <<<gV4, blk, 0, stream>>>(m, dinv, b + layer * D, agg, n);
    agg_edges<<<gEd, blk, 0, stream>>>(m, dinv, srcI, dstI, agg, E_);
    hipMemsetAsync(stats, 0, 2 * D * sizeof(float), stream);
    bn_partial <<<256, blk, 0, stream>>>(agg, stats, n);
    bn_finalize<<<1, dim3(D), 0, stream>>>(stats, gm + layer * D, bt + layer * D, n);
    bn_apply   <<<gV4, blk, 0, stream>>>(agg, stats, aP, layer, h, n);
    if (layer == 1)
      mask_dec<<<gND, blk, 0, stream>>>(mv, dec, h, n);
  }

  // final projection + bias + sigmoid, straight into d_out
  gemm_bf16_wmma<<<gGem, blk, 0, stream>>>(h, Wp, (float*)d_out, bp, n, 1);
}
